// SlowLSTM_33423435498395
// MI455X (gfx1250) — matrix operand order repaired and checked by reading, not compiled, not run on hardware
//
#include <hip/hip_runtime.h>
#include <hip/hip_bf16.h>

typedef __attribute__((ext_vector_type(16))) __bf16 v16bf;
typedef __attribute__((ext_vector_type(8)))  __bf16 v8bf;
typedef __attribute__((ext_vector_type(8)))  float  v8f;
typedef __attribute__((ext_vector_type(4)))  int    v4i;

#define B_DIM 4096
#define D_DIM 2048
#define H_DIM 2048
#define K_DIM 4096          // D + H concatenated
#define KSTEP 64            // staged K per step (2 WMMA K-chunks)
#define NSTEP (K_DIM / KSTEP)

// ---- CDNA5 async global->LDS copy (ASYNCcnt) with sync fallback ----------
#if __has_builtin(__builtin_amdgcn_global_load_async_to_lds_b128)
#define LSTM_ASYNC 1
#define COPY16(lp, gp)                                                      \
    __builtin_amdgcn_global_load_async_to_lds_b128(                         \
        (__attribute__((address_space(1))) v4i*)(gp),                       \
        (__attribute__((address_space(3))) v4i*)(lp), 0, 0)
#else
#define LSTM_ASYNC 0
#define COPY16(lp, gp) (*(v8bf*)(lp) = *(const v8bf*)(gp))
#endif

#if LSTM_ASYNC
#if __has_builtin(__builtin_amdgcn_s_wait_asynccnt)
#define WAIT_ASYNC() __builtin_amdgcn_s_wait_asynccnt(0)
#else
#define WAIT_ASYNC() asm volatile("s_wait_asynccnt 0x0" ::: "memory")
#endif
#else
#define WAIT_ASYNC() ((void)0)
#endif

// ---------------------------------------------------------------------------
// Pack [x | h] -> bf16 A[B][K], K-contiguous (row major).
// ---------------------------------------------------------------------------
__global__ void lstm_pack_a(const float* __restrict__ x,
                            const float* __restrict__ h,
                            __bf16* __restrict__ Abf) {
    int idx = blockIdx.x * blockDim.x + threadIdx.x;  // over B*K
    int b = idx >> 12;
    int k = idx & (K_DIM - 1);
    float v = (k < D_DIM) ? x[(size_t)b * D_DIM + k]
                          : h[(size_t)b * H_DIM + (k - D_DIM)];
    Abf[idx] = (__bf16)v;
}

// ---------------------------------------------------------------------------
// Pack gate weights -> bf16 Wt[4][H][K] (K-major, transposed vs source).
// gate order g: 0=i, 1=f, 2=o, 3=c ; k<D -> w_x, else w_h
// ---------------------------------------------------------------------------
__global__ void lstm_pack_w(const float* __restrict__ wxi, const float* __restrict__ whi,
                            const float* __restrict__ wxf, const float* __restrict__ whf,
                            const float* __restrict__ wxo, const float* __restrict__ who,
                            const float* __restrict__ wxc, const float* __restrict__ whc,
                            __bf16* __restrict__ Wt) {
    long idx = (long)blockIdx.x * blockDim.x + threadIdx.x;  // over 4*H*K
    int  k   = (int)(idx & (K_DIM - 1));
    long gn  = idx >> 12;
    int  n   = (int)(gn & (H_DIM - 1));
    int  g   = (int)(gn >> 11);

    const float* w;
    long off;
    if (k < D_DIM) {
        w   = (g == 0) ? wxi : (g == 1) ? wxf : (g == 2) ? wxo : wxc;
        off = (long)k * H_DIM + n;
    } else {
        w   = (g == 0) ? whi : (g == 1) ? whf : (g == 2) ? who : whc;
        off = (long)(k - D_DIM) * H_DIM + n;
    }
    Wt[idx] = (__bf16)w[off];
}

// ---------------------------------------------------------------------------
__device__ __forceinline__ float sigf(float x) {
    return 1.0f / (1.0f + __expf(-x));
}

// Block: 256 threads = 8 waves, tile 64(M) x 64(N), 4 gates.
// LDS double buffer: A 64x64 bf16 (8 KB) + W 4x64x64 bf16 (32 KB) per buffer.
// Per K-step: async-stage next 64-K tile, 16 v_wmma off current, wait+barrier.
__global__ __launch_bounds__(256)
void lstm_wmma_kernel(const __bf16* __restrict__ Abf,
                      const __bf16* __restrict__ Wt,
                      const float* __restrict__ c_in,
                      const float* __restrict__ b_i, const float* __restrict__ b_f,
                      const float* __restrict__ b_o, const float* __restrict__ b_c,
                      float* __restrict__ h_out, float* __restrict__ c_out) {
    __shared__ __bf16 sA[2][64 * KSTEP];       // 2 x 8 KB
    __shared__ __bf16 sW[2][4 * 64 * KSTEP];   // 2 x 32 KB

    const int tid   = threadIdx.x;
    const int lane  = tid & 31;
    const int wid   = tid >> 5;
    const int waveN = wid & 3;   // 4 waves across N (16 cols each)
    const int waveM = wid >> 2;  // 2 waves across M (32 rows each)
    const int l16   = lane & 15;
    const int hsel  = lane >> 4; // 0: lanes 0-15, 1: lanes 16-31

    const int rowB = blockIdx.y * 64;
    const int nB   = blockIdx.x * 64;

    // --- staging assignments (cooperative, 16B chunks) ---
    // A tile: 64 rows x 64 k = 512 chunks -> 2 per thread
    const __bf16* aSrc[2];
    int           aDst[2];
#pragma unroll
    for (int i = 0; i < 2; ++i) {
        int l   = tid + i * 256;
        int ch  = l & 7;
        int row = l >> 3;
        aSrc[i] = Abf + (size_t)(rowB + row) * K_DIM + ch * 8;
        aDst[i] = row * KSTEP + ch * 8;
    }

    // W tile: 4 gates x 64 n x 64 k = 2048 chunks -> 8 per thread
    const __bf16* wSrc[8];
    int           wDst[8];
#pragma unroll
    for (int i = 0; i < 8; ++i) {
        int l  = tid + i * 256;
        int ch = l & 7;
        int n  = (l >> 3) & 63;
        int g  = l >> 9;
        wSrc[i] = Wt + ((size_t)g * H_DIM + (nB + n)) * K_DIM + ch * 8;
        wDst[i] = (g * 64 + n) * KSTEP + ch * 8;
    }

    v8f acc[4][2];
#pragma unroll
    for (int g = 0; g < 4; ++g)
#pragma unroll
        for (int m = 0; m < 2; ++m)
            acc[g][m] = (v8f){0.f, 0.f, 0.f, 0.f, 0.f, 0.f, 0.f, 0.f};

    auto stage = [&](int k0, int buf) {
#pragma unroll
        for (int i = 0; i < 2; ++i)
            COPY16(&sA[buf][aDst[i]], aSrc[i] + k0);
#pragma unroll
        for (int i = 0; i < 8; ++i)
            COPY16(&sW[buf][wDst[i]], wSrc[i] + k0);
    };

    // Fragment bases in LDS.
    // A: lane m = l16; lanes 0-15 K 0-15,
    // lanes 16-31 K 16-31.  W: lane = n, K contiguous per half-wave.
    const int aOff = (waveM * 32 + l16) * KSTEP + hsel * 16;
    const int wOff = (waveN * 16 + l16) * KSTEP + hsel * 16;

    auto compute = [&](int buf) {
#pragma unroll
        for (int kc = 0; kc < KSTEP / 32; ++kc) {
            const int ko = kc * 32;
            // Load ALL fragments first (distinct regs) so the scheduler can
            // overlap ds_load latency instead of serializing on reuse.
            v16bf afr[2];
#pragma unroll
            for (int wm = 0; wm < 2; ++wm) {
                const __bf16* p = &sA[buf][aOff + wm * 16 * KSTEP + ko];
                v8bf lo = *(const v8bf*)(p);
                v8bf hi = *(const v8bf*)(p + 8);
                afr[wm] = __builtin_shufflevector(lo, hi, 0, 1, 2, 3, 4, 5, 6,
                                                  7, 8, 9, 10, 11, 12, 13, 14,
                                                  15);
            }
            v16bf bfr[4];
#pragma unroll
            for (int g = 0; g < 4; ++g)
                bfr[g] = *(const v16bf*)(&sW[buf][wOff + g * 64 * KSTEP + ko]);
#pragma unroll
            for (int g = 0; g < 4; ++g) {
                acc[g][0] = __builtin_amdgcn_wmma_f32_16x16x32_bf16(
                    false, afr[0], false, bfr[g], (short)0, acc[g][0], false,
                    false);
                acc[g][1] = __builtin_amdgcn_wmma_f32_16x16x32_bf16(
                    false, afr[1], false, bfr[g], (short)0, acc[g][1], false,
                    false);
            }
        }
    };

    // --- software pipeline: async DMA overlaps WMMA, 1 barrier per step ---
    stage(0, 0);
    WAIT_ASYNC();
    __syncthreads();
    int cur = 0;
#pragma unroll 1
    for (int s = 0; s < NSTEP; ++s) {
        if (s + 1 < NSTEP) stage((s + 1) * KSTEP, cur ^ 1);
        compute(cur);
        WAIT_ASYNC();
        __syncthreads();
        cur ^= 1;
    }

    // --- fused LSTM epilogue (C/D layout: VGPR r -> M=r / r+8) ---
    const int row0 = rowB + waveM * 32;
    const int col  = nB + waveN * 16 + l16;
    const float bi = b_i[col];
    const float bf = b_f[col];
    const float bo = b_o[col];
    const float bc = b_c[col];

#pragma unroll
    for (int wm = 0; wm < 2; ++wm) {
        const int rbase = row0 + wm * 16 + hsel * 8;
#pragma unroll
        for (int r = 0; r < 8; ++r) {
            const int row = rbase + r;
            const size_t o = (size_t)row * H_DIM + col;
            float iv = sigf(acc[0][wm][r] + bi);
            float fv = sigf(acc[1][wm][r] + bf);
            float ov = sigf(acc[2][wm][r] + bo);
            float gv = tanhf(acc[3][wm][r] + bc);
            float cn = c_in[o] * fv + iv * gv;
            float hn = ov * tanhf(cn);
            h_out[o] = hn;
            c_out[o] = cn;
        }
    }
}

// ---------------------------------------------------------------------------
extern "C" void kernel_launch(void* const* d_in, const int* in_sizes, int n_in,
                              void* d_out, int out_size, void* d_ws, size_t ws_size,
                              hipStream_t stream) {
    const float* x   = (const float*)d_in[0];
    const float* h   = (const float*)d_in[1];
    const float* c   = (const float*)d_in[2];
    const float* wxi = (const float*)d_in[3];
    const float* whi = (const float*)d_in[4];
    const float* wxf = (const float*)d_in[5];
    const float* whf = (const float*)d_in[6];
    const float* wxo = (const float*)d_in[7];
    const float* who = (const float*)d_in[8];
    const float* wxc = (const float*)d_in[9];
    const float* whc = (const float*)d_in[10];
    const float* b_i = (const float*)d_in[11];
    const float* b_f = (const float*)d_in[12];
    const float* b_o = (const float*)d_in[13];
    const float* b_c = (const float*)d_in[14];

    __bf16* Abf = (__bf16*)d_ws;                                      // 32 MB
    __bf16* Wt  = (__bf16*)((char*)d_ws + (size_t)B_DIM * K_DIM * 2); // 64 MB

    float* h_out = (float*)d_out;
    float* c_out = h_out + (size_t)B_DIM * H_DIM;

    lstm_pack_a<<<(B_DIM * K_DIM) / 256, 256, 0, stream>>>(x, h, Abf);

    long wElems = 4L * H_DIM * K_DIM;
    lstm_pack_w<<<(unsigned)(wElems / 256), 256, 0, stream>>>(
        wxi, whi, wxf, whf, wxo, who, wxc, whc, Wt);

    dim3 grid(H_DIM / 64, B_DIM / 64);
    lstm_wmma_kernel<<<grid, 256, 0, stream>>>(Abf, Wt, c, b_i, b_f, b_o, b_c,
                                               h_out, c_out);
}
